// ALEncoder_57028575756501
// MI455X (gfx1250) — matrix operand order repaired and checked by reading, not compiled, not run on hardware
//
#include <hip/hip_runtime.h>
#include <math.h>

// ---------------------------------------------------------------------------
// ALEncoder fused kernel for MI455X (gfx1250, wave32, WMMA 16x16x32 f16)
//
// Kernel 1: one block = (agent, lane-split). 128 threads = 4 waves.
//   streams 16-lane tiles: feature build -> 5x(Lin2->LN->ReLU in regs) ->
//   WMMA chain (5x W2 + aggrW + Wk + Wv, B operands register-resident f16
//   fragments) -> parallel online-softmax partials (m, s, acc) to workspace.
//   Row LayerNorm / softmax reductions use pure-VALU DPP lane swaps
//   (quad_perm xor1/xor2 + row_half_mirror + row_mirror) - no LDS crossbar.
//   The WMMA A staging buffer is double-buffered: one barrier per stage.
// Kernel 2: one block per agent; merges softmax partials, Wo / GRU gate /
//   LayerNorm / MLP tail, writes (N,64) f32 output.
// ---------------------------------------------------------------------------

typedef _Float16 v16h __attribute__((ext_vector_type(16)));
typedef _Float16 v8h  __attribute__((ext_vector_type(8)));
typedef float    v8f  __attribute__((ext_vector_type(8)));

#define EMBED   64
#define HEADS   8
#define NAGENT  64
#define NLANE   8192
#define SPLIT   16                      // lane-splits per agent -> 1024 blocks
#define TILES_PER_SPLIT ((NLANE / 16) / SPLIT)   // 32 tiles of 16 lanes
#define ABS     72                      // A-buffer row stride (f16), 144B = 16B aligned
#define NEGINF  (-3.0e38f)
#define LN_EPS  1e-5f

// DPP controls (wave32): quad_perm xor1 / xor2, half-row mirror (xor7 once
// quads are uniform), row mirror (xor15 once 8-groups are uniform).
#define DPP_XOR1   0xB1   // quad_perm [1,0,3,2]
#define DPP_XOR2   0x4E   // quad_perm [2,3,0,1]
#define DPP_HMIRR  0x141  // row_half_mirror
#define DPP_MIRR   0x140  // row_mirror

template <int CTRL>
__device__ inline float dpp_fadd(float x) {
    int v = __builtin_amdgcn_update_dpp(0, __float_as_int(x), CTRL, 0xF, 0xF, true);
    return x + __int_as_float(v);
}
template <int CTRL>
__device__ inline float dpp_fmax(float x) {
    int v = __builtin_amdgcn_update_dpp(0, __float_as_int(x), CTRL, 0xF, 0xF, true);
    return fmaxf(x, __int_as_float(v));
}
// Sum across 8-lane group (VALU only)
__device__ inline float sum8(float x) {
    x = dpp_fadd<DPP_XOR1>(x);
    x = dpp_fadd<DPP_XOR2>(x);
    x = dpp_fadd<DPP_HMIRR>(x);
    return x;
}
// Sum / max across 16-lane group (VALU only)
__device__ inline float sum16(float x) {
    x = dpp_fadd<DPP_XOR1>(x);
    x = dpp_fadd<DPP_XOR2>(x);
    x = dpp_fadd<DPP_HMIRR>(x);
    x = dpp_fadd<DPP_MIRR>(x);
    return x;
}
__device__ inline float max16(float x) {
    x = dpp_fmax<DPP_XOR1>(x);
    x = dpp_fmax<DPP_XOR2>(x);
    x = dpp_fmax<DPP_HMIRR>(x);
    x = dpp_fmax<DPP_MIRR>(x);
    return x;
}

struct P {
    const float* temporal;      // (64,64)
    const float* positions;     // (64,2,2)
    const float* lane_start;    // (8192,2)
    const float* lane_end;      // (8192,2)
    const int*   is_inter;      // (8192,)
    const int*   turn_dir;      // (8192,)
    const int*   traffic;       // (8192,)
    const float* W1[5];  const float* b1[5];
    const float* ln1w[5]; const float* ln1b[5];
    const float* W2[5];  const float* b2[5];
    const float* aggr_lnw;  const float* aggr_lnb;
    const float* aggr_W;    const float* aggr_b;
    const float* aggr_lnw2; const float* aggr_lnb2;
    const float* Wq; const float* Wk; const float* Wv; const float* Wo;
    const float* W_self; const float* b_self;
    const float* W_ih;   const float* b_ih;
    const float* W_hh;   const float* b_hh;
    const float* norm2w; const float* norm2b;
    const float* Wm1; const float* bm1;
    const float* Wm2; const float* bm2;
};

#define WMMA_F16(A, B, C) \
    __builtin_amdgcn_wmma_f32_16x16x32_f16(false, (A), false, (B), (short)0, (C), false, false)

// A operand (16x64 f16 row-major in LDS, stride ABS) -> 16x32 fragment kt in {0,1}.
// ISA layout: lanes 0-15 rows 0-15 hold K=kb..kb+7 (elems 0-7) and kb+16..kb+23
// (elems 8-15) with kb = 0; lanes 16-31 kb = 8.
__device__ inline v16h load_afrag(const _Float16* buf, int kt) {
    const int lane = threadIdx.x & 31;
    const int r    = lane & 15;
    const int base = ((lane >> 4) << 3) + (kt << 5);
    const _Float16* row = buf + r * ABS;
    v8h lo = *(const v8h*)(row + base);
    v8h hi = *(const v8h*)(row + base + 16);
    return __builtin_shufflevector(lo, hi, 0,1,2,3,4,5,6,7,8,9,10,11,12,13,14,15);
}

// B operand for Y = X @ W^T : B[k][c] = W[c][k]. Lane holds col = cbase+(lane&15),
// K = kb..kb+7 (elems 0-7) and kb+16..kb+23 (elems 8-15), kb = (lane>>4)*8 + kt*32.
__device__ inline v16h load_bfrag(const float* W, int cbase, int kt) {
    const int lane  = threadIdx.x & 31;
    const int col   = cbase + (lane & 15);
    const int kbase = ((lane >> 4) << 3) + (kt << 5);
    const float* row = W + col * EMBED + kbase;
    v16h b;
#pragma unroll
    for (int i = 0; i < 16; ++i) b[i] = (_Float16)row[i < 8 ? i : i + 8];
    return b;
}

// C/D fragment (f32 16x16) -> LDS f32 [16][64]: VGPR j holds row j + 8*(lane>>4),
// col = cbase + (lane&15).
__device__ inline void store_cfrag(float* buf, v8f c, int cbase) {
    const int lane = threadIdx.x & 31;
    const int col  = cbase + (lane & 15);
    const int r0   = (lane >> 4) << 3;
#pragma unroll
    for (int j = 0; j < 8; ++j) buf[(r0 + j) * EMBED + col] = c[j];
}

// Parallel row LayerNorm over 16 rows x 64 cols: thread t owns row r=t>>3,
// feature group g=t&7 (8 values, in regs). Mean/var via DPP sum8 within the
// 8-lane row group. Result (optionally ReLU'd, optionally masked) -> f16 A buf.
__device__ inline void ln_reduce_store(float v[8], const float* gamma,
                                       const float* beta, bool relu, float mk,
                                       _Float16* abuf) {
    const int t = threadIdx.x;
    const int r = t >> 3, g = t & 7;
    float s1 = 0.f;
#pragma unroll
    for (int i = 0; i < 8; ++i) s1 += v[i];
    s1 = sum8(s1);
    const float mu = s1 * (1.f / EMBED);
    float s2 = 0.f;
#pragma unroll
    for (int i = 0; i < 8; ++i) { const float d = v[i] - mu; s2 += d * d; }
    s2 = sum8(s2);
    const float rs = rsqrtf(s2 * (1.f / EMBED) + LN_EPS);
    v8h o;
#pragma unroll
    for (int i = 0; i < 8; ++i) {
        float y = (v[i] - mu) * rs * gamma[(g << 3) + i] + beta[(g << 3) + i];
        if (relu) y = fmaxf(y, 0.f);
        o[i] = (_Float16)(y * mk);
    }
    *(v8h*)(abuf + r * ABS + (g << 3)) = o;   // 16B-aligned f16x8 store
}

// Same, but source rows come from an LDS f32 [16][64] dump plus a bias vector.
__device__ inline void ln_from_tmp(const float* src, const float* bias,
                                   const float* gamma, const float* beta,
                                   bool relu, float mk, _Float16* abuf) {
    const int t = threadIdx.x;
    const int r = t >> 3, g = t & 7;
    const float* row = src + r * EMBED + (g << 3);
    float v[8];
#pragma unroll
    for (int i = 0; i < 8; ++i) v[i] = row[i] + bias[(g << 3) + i];
    ln_reduce_store(v, gamma, beta, relu, mk, abuf);
}

__launch_bounds__(128)
__global__ void alenc_lane_attn_kernel(P p, float* ws) {
    const int n     = blockIdx.x / SPLIT;   // agent
    const int split = blockIdx.x % SPLIT;
    const int t     = threadIdx.x;          // 0..127
    const int cbase = (t >> 5) << 4;        // wave's 16-column slab

    __shared__ __align__(16) _Float16 s_abuf[2][16 * ABS]; // double-buffered A staging
    __shared__ float s_tmp[16 * EMBED];                    // C-fragment dumps
    __shared__ float s_k[16 * EMBED];
    __shared__ float s_v[16 * EMBED];
    __shared__ float s_logit[HEADS][16];
    __shared__ float s_feat[16][8];
    __shared__ float s_mask[16];
    __shared__ float s_W1[5][64][2];
    __shared__ float s_b1[5][64], s_ln1w[5][64], s_ln1b[5][64];
    __shared__ float s_albw[64], s_albb[64], s_aggb[64], s_alw2[64], s_alb2[64];
    __shared__ float s_sumb2[64];
    __shared__ float s_q[64], s_temp[64];
    __shared__ float s_rot[4];
    __shared__ float s_m[8], s_s[8], s_sc[8], s_acc[64];

    // ---- per-block parameter / state setup ----
#pragma unroll
    for (int b = 0; b < 5; ++b) {
        ((float*)s_W1[b])[t] = p.W1[b][t];          // 128 values
        if (t < 64) {
            s_b1[b][t]   = p.b1[b][t];
            s_ln1w[b][t] = p.ln1w[b][t];
            s_ln1b[b][t] = p.ln1b[b][t];
        }
    }
    if (t < 64) {
        s_albw[t] = p.aggr_lnw[t];  s_albb[t] = p.aggr_lnb[t];
        s_aggb[t] = p.aggr_b[t];
        s_alw2[t] = p.aggr_lnw2[t]; s_alb2[t] = p.aggr_lnb2[t];
        float sb = 0.f;
#pragma unroll
        for (int b = 0; b < 5; ++b) sb += p.b2[b][t];
        s_sumb2[t] = sb;
        s_temp[t]  = p.temporal[n * EMBED + t];
        s_acc[t]   = 0.f;
    }
    if (t < 8) { s_m[t] = NEGINF; s_s[t] = 0.f; }
    if (t == 0) {
        float p0x = p.positions[n * 4 + 0], p0y = p.positions[n * 4 + 1];
        float p1x = p.positions[n * 4 + 2], p1y = p.positions[n * 4 + 3];
        float ang = atan2f(p1y - p0y, p1x - p0x);
        s_rot[0] = cosf(ang); s_rot[1] = sinf(ang);
        s_rot[2] = p1x;       s_rot[3] = p1y;
    }
    __syncthreads();

    if (t < 64) {   // q = temporal[n] @ Wq^T  (one 64-vec per agent)
        float acc = 0.f;
        for (int k = 0; k < EMBED; ++k) acc += s_temp[k] * p.Wq[t * EMBED + k];
        s_q[t] = acc;
    }

    // ---- register-resident B fragments (loop invariant) ----
    v16h fW2[5][2], fAg[2], fWk[2], fWv[2];
#pragma unroll
    for (int b = 0; b < 5; ++b) {
        fW2[b][0] = load_bfrag(p.W2[b], cbase, 0);
        fW2[b][1] = load_bfrag(p.W2[b], cbase, 1);
    }
    fAg[0] = load_bfrag(p.aggr_W, cbase, 0); fAg[1] = load_bfrag(p.aggr_W, cbase, 1);
    fWk[0] = load_bfrag(p.Wk,     cbase, 0); fWk[1] = load_bfrag(p.Wk,     cbase, 1);
    fWv[0] = load_bfrag(p.Wv,     cbase, 0); fWv[1] = load_bfrag(p.Wv,     cbase, 1);
    __syncthreads();

    const float rotc = s_rot[0], rots = s_rot[1], px = s_rot[2], py = s_rot[3];

    // ================= lane-tile loop =================
    for (int tt = 0; tt < TILES_PER_SPLIT; ++tt) {
        const int tile = split * TILES_PER_SPLIT + tt;

        // -- features for 16 lanes --
        if (t < 16) {
            const int li = tile * 16 + t;
            float sx = p.lane_start[2 * li], sy = p.lane_start[2 * li + 1];
            float ex = p.lane_end[2 * li],   ey = p.lane_end[2 * li + 1];
            float rx = sx - px, ry = sy - py;
            float vx = ex - sx, vy = ey - sy;
            s_feat[t][0] = rotc * vx - rots * vy;
            s_feat[t][1] = rots * vx + rotc * vy;
            s_feat[t][2] = rotc * rx - rots * ry;
            s_feat[t][3] = rots * rx + rotc * ry;
            s_feat[t][4] = (float)p.is_inter[li];
            s_feat[t][5] = (float)p.turn_dir[li];
            s_feat[t][6] = (float)p.traffic[li];
            s_mask[t] = (sqrtf(rx * rx + ry * ry) < 50.0f) ? 1.f : 0.f;
            const int nli = li + 16;
            if (nli < NLANE) {          // global_prefetch_b8 for next tile
                __builtin_prefetch(p.lane_start + 2 * nli, 0, 0);
                __builtin_prefetch(p.lane_end   + 2 * nli, 0, 0);
            }
        }
        __syncthreads();

        // -- 5 branches: Lin(2->64)+LN+ReLU in regs -> WMMA W2 accumulate.
        //    A-buffer ping-pongs (b&1): one barrier per branch; the buffer
        //    being overwritten was last read before the previous barrier.
        v8f ctot = {0.f,0.f,0.f,0.f,0.f,0.f,0.f,0.f};
#pragma unroll
        for (int b = 0; b < 5; ++b) {
            {   // thread t: row r=t>>3, features g*8..g*8+7 (same map as LN)
                const int r = t >> 3, jb = (t & 7) << 3;
                float x0, x1;
                if      (b == 0) { x0 = s_feat[r][0]; x1 = s_feat[r][1]; }
                else if (b == 1) { x0 = s_feat[r][2]; x1 = s_feat[r][3]; }
                else             { x0 = s_feat[r][4 + (b - 2)]; x1 = x0; }
                float v[8];
#pragma unroll
                for (int j = 0; j < 8; ++j) {
                    const int jj = jb + j;
                    v[j] = x0 * s_W1[b][jj][0] + x1 * s_W1[b][jj][1] + s_b1[b][jj];
                }
                ln_reduce_store(v, s_ln1w[b], s_ln1b[b], true, 1.f, s_abuf[b & 1]);
            }
            __syncthreads();
            ctot = WMMA_F16(load_afrag(s_abuf[b & 1], 0), fW2[b][0], ctot);
            ctot = WMMA_F16(load_afrag(s_abuf[b & 1], 1), fW2[b][1], ctot);
        }

        // -- aggregate: LN(total+sum_b2) -> ReLU -> aggrW -> (+b) LN2 -> mask --
        store_cfrag(s_tmp, ctot, cbase);        // (branch4 read buf0)
        __syncthreads();
        ln_from_tmp(s_tmp, s_sumb2, s_albw, s_albb, true, 1.f, s_abuf[1]);
        __syncthreads();
        v8f cag = {0.f,0.f,0.f,0.f,0.f,0.f,0.f,0.f};
        cag = WMMA_F16(load_afrag(s_abuf[1], 0), fAg[0], cag);
        cag = WMMA_F16(load_afrag(s_abuf[1], 1), fAg[1], cag);
        store_cfrag(s_tmp, cag, cbase);
        __syncthreads();
        ln_from_tmp(s_tmp, s_aggb, s_alw2, s_alb2, false, s_mask[t >> 3], s_abuf[0]);
        __syncthreads();

        // -- k, v projections --
        {
            v16h a0 = load_afrag(s_abuf[0], 0), a1 = load_afrag(s_abuf[0], 1);
            v8f ck = {0.f,0.f,0.f,0.f,0.f,0.f,0.f,0.f};
            ck = WMMA_F16(a0, fWk[0], ck); ck = WMMA_F16(a1, fWk[1], ck);
            store_cfrag(s_k, ck, cbase);
            v8f cv = {0.f,0.f,0.f,0.f,0.f,0.f,0.f,0.f};
            cv = WMMA_F16(a0, fWv[0], cv); cv = WMMA_F16(a1, fWv[1], cv);
            store_cfrag(s_v, cv, cbase);
        }
        __syncthreads();

        // -- attention logits + online softmax, fully parallel (VALU DPP) --
        {   // head h = t>>4 (2 heads per wave), row r = t&15
            const int h = t >> 4, r = t & 15;
            float acc = 0.f;
#pragma unroll
            for (int d = 0; d < 8; ++d)
                acc += s_q[h * 8 + d] * s_k[r * EMBED + h * 8 + d];
            const float lg = (s_mask[r] != 0.f) ? acc * 0.35355339059f : NEGINF;
            const float tm = max16(lg);         // max over the 16 rows of head h
            const float mo = s_m[h];            // read before any lane writes
            const float nm = fmaxf(mo, tm);
            const float sc = (nm < -1e37f) ? 1.f : __expf(mo - nm);
            const float w  = (nm < -1e37f || lg < -1e37f) ? 0.f : __expf(lg - nm);
            s_logit[h][r] = w;
            const float ss = sum16(w);          // sum over the 16 rows of head h
            if (r == 0) { s_s[h] = s_s[h] * sc + ss; s_m[h] = nm; s_sc[h] = sc; }
        }
        __syncthreads();
        if (t < 64) {                   // acc[h][d] rescale + accumulate
            const int h = t >> 3, d = t & 7;
            float a = s_acc[t] * s_sc[h];
#pragma unroll
            for (int r = 0; r < 16; ++r)
                a += s_logit[h][r] * s_v[r * EMBED + h * 8 + d];
            s_acc[t] = a;
        }
        __syncthreads();
    }

    // ---- write softmax partials ----
    float* wsp = ws + (size_t)(n * SPLIT + split) * 80;
    if (t < 8)  { wsp[t] = s_m[t]; wsp[8 + t] = s_s[t]; }
    if (t < 64) wsp[16 + t] = s_acc[t];
}

__launch_bounds__(128)
__global__ void alenc_finalize_kernel(P p, const float* ws, float* out) {
    const int n = blockIdx.x;
    const int t = threadIdx.x;

    __shared__ float m[8], sden[8], temp[64], msg[64], nmv[64], outv[64], hn[64];
    __shared__ float hid[256], stat[2];

    if (t < 8) {
        float M = NEGINF;
        for (int sp = 0; sp < SPLIT; ++sp)
            M = fmaxf(M, ws[(size_t)(n * SPLIT + sp) * 80 + t]);
        m[t] = M;
    }
    if (t < 64) temp[t] = p.temporal[n * EMBED + t];
    __syncthreads();
    if (t < 8) {
        float S = 0.f;
        for (int sp = 0; sp < SPLIT; ++sp) {
            const float* w = ws + (size_t)(n * SPLIT + sp) * 80;
            S += w[8 + t] * __expf(w[t] - m[t]);
        }
        sden[t] = S;
    }
    __syncthreads();
    if (t < 64) {                       // merge acc partials, msg = acc/s
        const int h = t >> 3;
        float A = 0.f;
        for (int sp = 0; sp < SPLIT; ++sp) {
            const float* w = ws + (size_t)(n * SPLIT + sp) * 80;
            A += w[16 + t] * __expf(w[h] - m[h]);
        }
        msg[t] = (sden[h] > 1e-30f) ? A / sden[h] : 0.f;
    }
    __syncthreads();
    if (t < 64) {                       // new_msg = msg @ Wo^T
        float a = 0.f;
        for (int k = 0; k < EMBED; ++k) a += p.Wo[t * EMBED + k] * msg[k];
        nmv[t] = a;
    }
    __syncthreads();
    if (t < 64) {                       // GRU-style gated fusion
        float g = p.b_ih[t], g2 = p.b_hh[t], sv = p.b_self[t];
        for (int k = 0; k < EMBED; ++k) {
            g  += p.W_ih[t * EMBED + k]   * nmv[k];
            g2 += p.W_hh[t * EMBED + k]   * temp[k];
            sv += p.W_self[t * EMBED + k] * temp[k];
        }
        const float gate = 1.f / (1.f + __expf(-(g + g2)));
        outv[t] = nmv[t] + gate * (sv - nmv[t]);
    }
    __syncthreads();
    if (t == 0) {                       // LayerNorm stats for norm2
        float mu = 0.f;
        for (int j = 0; j < EMBED; ++j) mu += outv[j];
        mu *= (1.f / EMBED);
        float var = 0.f;
        for (int j = 0; j < EMBED; ++j) { float d = outv[j] - mu; var += d * d; }
        stat[0] = mu;
        stat[1] = rsqrtf(var * (1.f / EMBED) + LN_EPS);
    }
    __syncthreads();
    if (t < 64) hn[t] = (outv[t] - stat[0]) * stat[1] * p.norm2w[t] + p.norm2b[t];
    __syncthreads();
    for (int f = t; f < 256; f += 128) { // MLP hidden 64 -> 256
        float a = p.bm1[f];
        for (int k = 0; k < EMBED; ++k) a += p.Wm1[f * EMBED + k] * hn[k];
        hid[f] = fmaxf(a, 0.f);
    }
    __syncthreads();
    if (t < 64) {                       // residual out 256 -> 64
        float a = p.bm2[t];
        for (int f = 0; f < 256; ++f) a += p.Wm2[t * 256 + f] * hid[f];
        out[n * EMBED + t] = outv[t] + a;
    }
}

extern "C" void kernel_launch(void* const* d_in, const int* in_sizes, int n_in,
                              void* d_out, int out_size, void* d_ws, size_t ws_size,
                              hipStream_t stream) {
    (void)in_sizes; (void)n_in; (void)out_size; (void)ws_size;
    // Leaf order: setup_inputs() dict insertion order, params flattened
    // recursively (branches list then each branch dict in insertion order).
    P p;
    p.temporal   = (const float*)d_in[0];
    p.positions  = (const float*)d_in[1];
    p.lane_start = (const float*)d_in[2];
    p.lane_end   = (const float*)d_in[3];
    p.is_inter   = (const int*)d_in[4];
    p.turn_dir   = (const int*)d_in[5];
    p.traffic    = (const int*)d_in[6];
    int i = 7;
    for (int b = 0; b < 5; ++b) {
        p.W1[b]   = (const float*)d_in[i++];
        p.b1[b]   = (const float*)d_in[i++];
        p.ln1w[b] = (const float*)d_in[i++];
        p.ln1b[b] = (const float*)d_in[i++];
        p.W2[b]   = (const float*)d_in[i++];
        p.b2[b]   = (const float*)d_in[i++];
    }
    p.aggr_lnw  = (const float*)d_in[i++];
    p.aggr_lnb  = (const float*)d_in[i++];
    p.aggr_W    = (const float*)d_in[i++];
    p.aggr_b    = (const float*)d_in[i++];
    p.aggr_lnw2 = (const float*)d_in[i++];
    p.aggr_lnb2 = (const float*)d_in[i++];
    p.Wq = (const float*)d_in[i++];
    p.Wk = (const float*)d_in[i++];
    p.Wv = (const float*)d_in[i++];
    p.Wo = (const float*)d_in[i++];
    p.W_self = (const float*)d_in[i++];
    p.b_self = (const float*)d_in[i++];
    p.W_ih   = (const float*)d_in[i++];
    p.b_ih   = (const float*)d_in[i++];
    p.W_hh   = (const float*)d_in[i++];
    p.b_hh   = (const float*)d_in[i++];
    p.norm2w = (const float*)d_in[i++];
    p.norm2b = (const float*)d_in[i++];
    p.Wm1 = (const float*)d_in[i++];
    p.bm1 = (const float*)d_in[i++];
    p.Wm2 = (const float*)d_in[i++];
    p.bm2 = (const float*)d_in[i++];

    float* ws = (float*)d_ws;   // 64*16*80 floats = 320 KB of partials
    alenc_lane_attn_kernel<<<NAGENT * SPLIT, 128, 0, stream>>>(p, ws);
    alenc_finalize_kernel<<<NAGENT, 128, 0, stream>>>(p, ws, (float*)d_out);
}
